// PairwiseConv1D_19842748907585
// MI455X (gfx1250) — matrix operand order repaired and checked by reading, not compiled, not run on hardware
//
#include <hip/hip_runtime.h>

// CDNA5 / gfx1250: conv1d-as-GEMM via bf16x3 split precision.
// out[b,i,f] = sum_{j<896} x_b[i*128+j] * w[j*128+f]
// x split online into (hi,lo) bf16; w pre-split + fragment-swizzled into d_ws.
// a*b ~= al*bh + ah*bl + ah*bh, f32 accumulation, V_WMMA_F32_16X16X32_BF16.
// Each wave owns TWO 16-row M-tiles (M=32) x all 128 N, so every B fragment
// loaded feeds 6 WMMAs.

typedef __attribute__((ext_vector_type(4)))  float  v4f;
typedef __attribute__((ext_vector_type(8)))  float  v8f;
typedef __attribute__((ext_vector_type(16))) __bf16 v16bf;

namespace {
constexpr int kB    = 32;
constexpr int kL    = 8192;
constexpr int kC    = 128;
constexpr int kK    = 7;
constexpr int kF    = 128;
constexpr int kLout = kL - kK + 1;                 // 8186
constexpr int kKred = kK * kC;                     // 896 (GEMM K)
constexpr int kChunks = kKred / 32;                // 28 K-chunks of 32
constexpr int kMPerWave = 32;                      // two 16-row WMMA tiles
constexpr int kTilesPerBatch = (kLout + kMPerWave - 1) / kMPerWave;  // 256
constexpr int kWavesPerBlock = 8;                  // 256 threads, wave32
constexpr int kWElems = kKred * kF;                // 114688 bf16 per plane
}

static __device__ __forceinline__ unsigned short f32_bf16_rn(float f) {
  union { float f; unsigned u; } v; v.f = f;
  unsigned u = v.u;
  u += 0x7FFFu + ((u >> 16) & 1u);                 // round-to-nearest-even
  return (unsigned short)(u >> 16);
}
static __device__ __forceinline__ float bf16up(unsigned short h) {
  union { unsigned u; float f; } v; v.u = ((unsigned)h) << 16;
  return v.f;
}

// ---- prep: split w[896][128] into hi/lo bf16 planes, fragment-swizzled ----
// B-frag layout (16-bit 32x16): elem t<8 <-> K=half*8+t,
// elem t>=8 <-> K=16+half*8+(t-8). ws index = ((c*8 + n)*32 + lane)*16 + t.
__global__ __launch_bounds__(256)
void PairwiseConv1D_prep_w(const float* __restrict__ w,
                           unsigned short* __restrict__ whi,
                           unsigned short* __restrict__ wlo) {
  const int idx = (int)blockIdx.x * 256 + (int)threadIdx.x;
  if (idx >= kWElems) return;
  const int t    = idx & 15;
  const int lane = (idx >> 4) & 31;
  const int n    = (idx >> 9) & 7;
  const int c    = idx >> 12;
  const int Kd   = 32 * c + ((lane >> 4) << 3) + (t & 7) + ((t >> 3) << 4);   // GEMM K index
  const int N    = 16 * n + (lane & 15);              // F index
  const float f  = w[(size_t)Kd * kF + N];
  const unsigned short h = f32_bf16_rn(f);
  whi[idx] = h;
  wlo[idx] = f32_bf16_rn(f - bf16up(h));
}

// ---- main GEMM: each wave owns a 32(M) x 128(N) output tile ----
__global__ __launch_bounds__(256)
void PairwiseConv1D_19842748907585_kernel(const float* __restrict__ x,
                                          const unsigned short* __restrict__ whi,
                                          const unsigned short* __restrict__ wlo,
                                          float* __restrict__ out) {
  const int lane = (int)(threadIdx.x & 31u);
  const int wave = (int)(threadIdx.x >> 5u);
  const int tile = (int)blockIdx.x * kWavesPerBlock + wave;
  const int b    = tile / kTilesPerBatch;
  const int i0   = (tile - b * kTilesPerBatch) * kMPerWave;

  const int half = lane >> 4;
  const int l15  = lane & 15;

  // A rows (contiguous im2col); clamp tail rows (discarded at store).
  int arow0 = i0 + l15;
  int arow1 = i0 + 16 + l15;
  if (arow0 >= kLout) arow0 = kLout - 1;
  if (arow1 >= kLout) arow1 = kLout - 1;
  // 16-bit A 16x32 layout: elems 0..7  <-> K = half*8 + {0..7}
  //                        elems 8..15 <-> K = 16 + half*8 + {0..7}
  const float* aptr0 = x + (size_t)b * (kL * kC) + (size_t)arow0 * kC + half * 8;
  const float* aptr1 = x + (size_t)b * (kL * kC) + (size_t)arow1 * kC + half * 8;

  const v16bf* bh = (const v16bf*)whi;
  const v16bf* bl = (const v16bf*)wlo;

  v8f acc0[8], acc1[8];
  const v8f zero = {};
#pragma unroll
  for (int n = 0; n < 8; ++n) { acc0[n] = zero; acc1[n] = zero; }

  for (int c = 0; c < kChunks; ++c) {
    const float* ap0 = aptr0 + c * 32;
    const float* ap1 = aptr1 + c * 32;
    v4f g0[4], g1[4];
    g0[0] = *(const v4f*)(ap0);      g0[1] = *(const v4f*)(ap0 + 4);
    g0[2] = *(const v4f*)(ap0 + 16); g0[3] = *(const v4f*)(ap0 + 20);
    g1[0] = *(const v4f*)(ap1);      g1[1] = *(const v4f*)(ap1 + 4);
    g1[2] = *(const v4f*)(ap1 + 16); g1[3] = *(const v4f*)(ap1 + 20);
    __builtin_prefetch(ap0 + 256, 0, 0);   // stream x ahead
    __builtin_prefetch(ap1 + 256, 0, 0);

    // hi/lo bf16 split with native converts (RNE)
    v16bf ah0, al0, ah1, al1;
#pragma unroll
    for (int e = 0; e < 16; ++e) {
      {
        const float f = g0[e >> 2][e & 3];
        const __bf16 h = (__bf16)f;
        ah0[e] = h;
        al0[e] = (__bf16)(f - (float)h);
      }
      {
        const float f = g1[e >> 2][e & 3];
        const __bf16 h = (__bf16)f;
        ah1[e] = h;
        al1[e] = (__bf16)(f - (float)h);
      }
    }

    const int fbase = (c * 8) * 32 + lane;
#pragma unroll
    for (int n = 0; n < 8; ++n) {
      const v16bf bhi = bh[fbase + n * 32];   // 2x global_load_b128
      const v16bf blo = bl[fbase + n * 32];
      // small terms first, then dominant hi*hi; B frag feeds 6 WMMAs
      acc0[n] = __builtin_amdgcn_wmma_f32_16x16x32_bf16(
          false, al0, false, bhi, (short)0, acc0[n], false, false);
      acc1[n] = __builtin_amdgcn_wmma_f32_16x16x32_bf16(
          false, al1, false, bhi, (short)0, acc1[n], false, false);
      acc0[n] = __builtin_amdgcn_wmma_f32_16x16x32_bf16(
          false, ah0, false, blo, (short)0, acc0[n], false, false);
      acc1[n] = __builtin_amdgcn_wmma_f32_16x16x32_bf16(
          false, ah1, false, blo, (short)0, acc1[n], false, false);
      acc0[n] = __builtin_amdgcn_wmma_f32_16x16x32_bf16(
          false, ah0, false, bhi, (short)0, acc0[n], false, false);
      acc1[n] = __builtin_amdgcn_wmma_f32_16x16x32_bf16(
          false, ah1, false, bhi, (short)0, acc1[n], false, false);
    }
  }

  // C/D layout: VGPR v, lanes 0-15 -> M=v ; lanes 16-31 -> M=v+8 ; N=l15
#pragma unroll
  for (int v = 0; v < 8; ++v) {
    const int row0 = i0 + 8 * half + v;
    if (row0 < kLout) {
      float* op = out + ((size_t)b * kLout + row0) * kF + l15;
#pragma unroll
      for (int n = 0; n < 8; ++n) op[n * 16] = acc0[n][v];
    }
    const int row1 = i0 + 16 + 8 * half + v;
    if (row1 < kLout) {
      float* op = out + ((size_t)b * kLout + row1) * kF + l15;
#pragma unroll
      for (int n = 0; n < 8; ++n) op[n * 16] = acc1[n][v];
    }
  }
}

extern "C" void kernel_launch(void* const* d_in, const int* in_sizes, int n_in,
                              void* d_out, int out_size, void* d_ws, size_t ws_size,
                              hipStream_t stream) {
  const float* x = (const float*)d_in[0];   // [32, 8192, 128, 1] f32
  const float* w = (const float*)d_in[1];   // [7, 128, 128]     f32
  float* out = (float*)d_out;               // [32, 8186, 128, 1] f32

  unsigned short* whi = (unsigned short*)d_ws;          // 229376 B
  unsigned short* wlo = whi + kWElems;                  // 229376 B (459 KB total)

  PairwiseConv1D_prep_w<<<(kWElems + 255) / 256, 256, 0, stream>>>(w, whi, wlo);

  const int totalTiles = kB * kTilesPerBatch;           // 8192 M-tiles (M=32)
  const int blocks = totalTiles / kWavesPerBlock;       // 1024
  PairwiseConv1D_19842748907585_kernel<<<blocks, 32 * kWavesPerBlock, 0, stream>>>(
      x, whi, wlo, out);
}
